// HAM_86079734547101
// MI455X (gfx1250) — matrix operand order repaired and checked by reading, not compiled, not run on hardware
//
#include <hip/hip_runtime.h>
#include <hip/hip_bf16.h>

typedef __bf16 bf16_t;
typedef __attribute__((ext_vector_type(16))) __bf16 v16bf;
typedef __attribute__((ext_vector_type(8)))  __bf16 v8bf;
typedef __attribute__((ext_vector_type(8)))  float  v8f;

// ---------------- helpers ----------------

__device__ __forceinline__ v8f v8f_zero() {
  v8f z;
#pragma unroll
  for (int i = 0; i < 8; ++i) z[i] = 0.f;
  return z;
}

__device__ __forceinline__ float sigm(float x) { return 1.f / (1.f + __expf(-x)); }

__device__ __forceinline__ float tanh_f(float x) {
  float e = __expf(-2.f * fabsf(x));
  float t = (1.f - e) / (1.f + e);
  return x >= 0.f ? t : -t;
}

// WMMA D = A(16x32 bf16) * B(32x16 bf16) + C (f32)
__device__ __forceinline__ v8f wmma_bf16(v16bf a, v16bf b, v8f c) {
  return __builtin_amdgcn_wmma_f32_16x16x32_bf16(false, a, false, b, (short)0, c,
                                                 false, false);
}

// A fragment: 16x32 bf16 block at column k0 of a row-major MxK matrix (ld elems).
// Lane l<16 : row m=l,   K = 0..15  of the block
// Lane l>=16: row m=l-16,K = 16..31 of the block
__device__ __forceinline__ v16bf load_fragA(const bf16_t* src, int ld, int k0) {
  const int lane = threadIdx.x & 31;
  const int m = lane & 15;
  const int ko = (lane >> 4) * 16;
  const bf16_t* p = src + m * ld + k0 + ko;
  v8bf lo = *(const v8bf*)(p);
  v8bf hi = *(const v8bf*)(p + 8);
  v16bf r;
#pragma unroll
  for (int i = 0; i < 8; ++i) { r[i] = lo[i]; r[i + 8] = hi[i]; }
  return r;
}

// B fragment: B[k][j] = W[j0+j][k0+k], W row-major (ldw elems per row).
// Lane l<16: col j=l, K=0..15 ; lane l>=16: col j=l-16, K=16..31 (contiguous in W row)
__device__ __forceinline__ v16bf load_fragB(const bf16_t* W, int ldw, int j0, int k0) {
  const int lane = threadIdx.x & 31;
  const int j = lane & 15;
  const int ko = (lane >> 4) * 16;
  return *(const v16bf*)(W + (size_t)(j0 + j) * ldw + k0 + ko);
}

// D fragment store: 16x16 f32 tile to row-major dst (ld elems), cols [j0, j0+16)
__device__ __forceinline__ void store_fragD(float* dst, int ld, int j0, v8f d) {
  const int lane = threadIdx.x & 31;
  const int col = j0 + (lane & 15);
  const int r0 = (lane >> 4) * 8;
#pragma unroll
  for (int i = 0; i < 8; ++i) dst[(r0 + i) * ld + col] = d[i];
}

// ---------------- prep: fold emb@wih^T+bih into byte tables, bf16 weight copies ----

__global__ __launch_bounds__(256) void prep_kernel(
    const float* __restrict__ emb,
    const float* __restrict__ wih_f, const float* __restrict__ bih_f,
    const float* __restrict__ wih_b, const float* __restrict__ bih_b,
    const float* __restrict__ whh_f, const float* __restrict__ whh_b,
    const float* __restrict__ pwhh_f, const float* __restrict__ pwhh_b,
    const float* __restrict__ pwih_f, const float* __restrict__ pwih_b,
    const float* __restrict__ ba_w1,
    bf16_t* __restrict__ table_f, bf16_t* __restrict__ table_b,
    bf16_t* __restrict__ whhf_bf, bf16_t* __restrict__ whhb_bf,
    bf16_t* __restrict__ pwhhf_bf, bf16_t* __restrict__ pwhhb_bf,
    bf16_t* __restrict__ pwihf_bf, bf16_t* __restrict__ pwihb_bf,
    bf16_t* __restrict__ baw1_bf) {
  const int job = blockIdx.y;
  const int idx = blockIdx.x * 256 + threadIdx.x;
  if (job <= 1) {
    if (idx < 256 * 384) {
      const int v = idx / 384, g = idx % 384;
      const float* w = job ? wih_b : wih_f;
      const float* bi = job ? bih_b : bih_f;
      float acc = bi[g];
#pragma unroll 8
      for (int e = 0; e < 64; ++e) acc += emb[v * 64 + e] * w[g * 64 + e];
      (job ? table_b : table_f)[idx] = (bf16_t)acc;
    }
  } else if (job <= 5) {
    if (idx < 384 * 128) {
      const float* s = (job == 2) ? whh_f : (job == 3) ? whh_b : (job == 4) ? pwhh_f : pwhh_b;
      bf16_t* d = (job == 2) ? whhf_bf : (job == 3) ? whhb_bf : (job == 4) ? pwhhf_bf : pwhhb_bf;
      d[idx] = (bf16_t)s[idx];
    }
  } else if (job <= 7) {
    if (idx < 384 * 256) {
      const float* s = (job == 6) ? pwih_f : pwih_b;
      bf16_t* d = (job == 6) ? pwihf_bf : pwihb_bf;
      d[idx] = (bf16_t)s[idx];
    }
  } else {
    if (idx < 128 * 256) baw1_bf[idx] = (bf16_t)ba_w1[idx];
  }
}

// ---------------- byte-level GRU (one direction per blockIdx.y) ----------------
// grid (128,2), 256 threads = 8 waves; each block: 16 batch rows, full H=128.
// Per step: hg(16x384) = h(16x128) @ whh^T via WMMA; gates on VALU; h in regs.

__global__ __launch_bounds__(256) void byte_gru_kernel(
    const long long* __restrict__ flow,
    const bf16_t* __restrict__ table_f, const bf16_t* __restrict__ table_b,
    const bf16_t* __restrict__ whhf, const bf16_t* __restrict__ whhb,
    const float* __restrict__ bhhf, const float* __restrict__ bhhb,
    bf16_t* __restrict__ out_bytes) {
  const int dir = blockIdx.y;
  const bf16_t* table = dir ? table_b : table_f;
  const bf16_t* whh_g = dir ? whhb : whhf;
  const float* bhh = dir ? bhhb : bhhf;

  __shared__ __align__(32) bf16_t whh_lds[384 * 128];
  __shared__ __align__(16) bf16_t h_bf[16 * 128];
  __shared__ float hg_lds[16 * 384];

  const int tid = threadIdx.x;
  {  // stage whh (96KB) into LDS, zero h
    const uint4* src = (const uint4*)whh_g;
    uint4* dst = (uint4*)whh_lds;
    for (int i = tid; i < 6144; i += 256) dst[i] = src[i];
    for (int i = tid; i < 1024; i += 256) ((unsigned*)h_bf)[i] = 0u;
  }
  __syncthreads();

  const int w = tid >> 5;  // wave id: owns gate-column tiles [3w, 3w+3)
  v16bf Bf[3][4];
#pragma unroll
  for (int n = 0; n < 3; ++n)
#pragma unroll
    for (int s = 0; s < 4; ++s)
      Bf[n][s] = load_fragB(whh_lds, 128, (w * 3 + n) * 16, s * 32);

  const int m = tid >> 4;          // gate-math row 0..15
  const int j0 = (tid & 15) * 8;   // gate-math col chunk
  const int nglob = blockIdx.x * 16 + m;
  float bhr[8], bhz[8], bhn[8], hreg[8];
#pragma unroll
  for (int i = 0; i < 8; ++i) {
    bhr[i] = bhh[j0 + i];
    bhz[i] = bhh[128 + j0 + i];
    bhn[i] = bhh[256 + j0 + i];
    hreg[i] = 0.f;
  }

  for (int t = 0; t < 128; ++t) {
    v8f a0 = v8f_zero(), a1 = v8f_zero(), a2 = v8f_zero();
#pragma unroll
    for (int s = 0; s < 4; ++s) {
      v16bf a = load_fragA(h_bf, 128, s * 32);
      a0 = wmma_bf16(a, Bf[0][s], a0);
      a1 = wmma_bf16(a, Bf[1][s], a1);
      a2 = wmma_bf16(a, Bf[2][s], a2);
    }
    store_fragD(hg_lds, 384, (w * 3 + 0) * 16, a0);
    store_fragD(hg_lds, 384, (w * 3 + 1) * 16, a1);
    store_fragD(hg_lds, 384, (w * 3 + 2) * 16, a2);
    __syncthreads();

    const int tt = dir ? (127 - t) : t;
    const int byte = (int)(flow[(size_t)nglob * 128 + tt]) & 255;
    const bf16_t* tr = table + byte * 384;  // xg row (bih folded in)
    v8bf hob;
#pragma unroll
    for (int i = 0; i < 8; ++i) {
      const int j = j0 + i;
      float xr = (float)tr[j], xz = (float)tr[128 + j], xn = (float)tr[256 + j];
      float hr = hg_lds[m * 384 + j] + bhr[i];
      float hz = hg_lds[m * 384 + 128 + j] + bhz[i];
      float hn = hg_lds[m * 384 + 256 + j] + bhn[i];
      float r = sigm(xr + hr);
      float z = sigm(xz + hz);
      float nn = tanh_f(xn + r * hn);
      float h = (1.f - z) * nn + z * hreg[i];
      hreg[i] = h;
      bf16_t hb = (bf16_t)h;
      h_bf[m * 128 + j] = hb;
      hob[i] = hb;
    }
    *(v8bf*)(out_bytes + (((size_t)tt * 2048 + nglob) * 256 + dir * 128 + j0)) = hob;
    __syncthreads();
  }
}

// ---------------- byte attention scores: tanh(out @ ba_w1^T + b1) @ ba_w2 + b2 ----

__global__ __launch_bounds__(256) void byte_attn_kernel(
    const bf16_t* __restrict__ outb, const bf16_t* __restrict__ baw1,
    const float* __restrict__ ba_b1, const float* __restrict__ ba_w2,
    const float* __restrict__ ba_b2, float* __restrict__ scores) {
  __shared__ float H[16 * 128];
  __shared__ float red[256];
  const int tid = threadIdx.x;
  const int w = tid >> 5;
  const size_t r0 = (size_t)blockIdx.x * 16;
  v8f acc = v8f_zero();
#pragma unroll
  for (int s = 0; s < 8; ++s) {
    v16bf a = load_fragA(outb + r0 * 256, 256, s * 32);
    v16bf b = load_fragB(baw1, 256, w * 16, s * 32);
    acc = wmma_bf16(a, b, acc);
  }
  store_fragD(H, 128, w * 16, acc);
  __syncthreads();
  const int m = tid >> 4, j0 = (tid & 15) * 8;
  float part = 0.f;
#pragma unroll
  for (int i = 0; i < 8; ++i) {
    const int j = j0 + i;
    part += tanh_f(H[m * 128 + j] + ba_b1[j]) * ba_w2[j];
  }
  red[tid] = part;
  __syncthreads();
  if (tid < 16) {
    float s = ba_b2[0];
#pragma unroll
    for (int q = 0; q < 16; ++q) s += red[tid * 16 + q];
    scores[r0 + tid] = s;
  }
}

// softmax over b (64) for each (l,p); in place
__global__ __launch_bounds__(64) void byte_softmax_kernel(float* __restrict__ scores) {
  __shared__ float s[64];
  const int g = blockIdx.x;  // l*32 + p
  const int l = g >> 5, p = g & 31;
  const int tid = threadIdx.x;
  const int idx = l * 2048 + p * 64 + tid;
  const float v = scores[idx];
  s[tid] = v;
  __syncthreads();
  for (int o = 32; o >= 1; o >>= 1) {
    if (tid < o) s[tid] = fmaxf(s[tid], s[tid + o]);
    __syncthreads();
  }
  const float mx = s[0];
  __syncthreads();
  const float e = __expf(v - mx);
  s[tid] = e;
  __syncthreads();
  for (int o = 32; o >= 1; o >>= 1) {
    if (tid < o) s[tid] += s[tid + o];
    __syncthreads();
  }
  scores[idx] = e / s[0];
}

// packet_emb[n][d] = sum_l out[l][n][d] * attn[l][n]
__global__ __launch_bounds__(256) void packet_emb_kernel(
    const bf16_t* __restrict__ outb, const float* __restrict__ attn,
    float* __restrict__ pe) {
  const int n = blockIdx.x, d = threadIdx.x;
  float acc = 0.f;
  for (int l = 0; l < 128; ++l)
    acc += (float)outb[((size_t)l * 2048 + n) * 256 + d] * attn[l * 2048 + n];
  pe[n * 256 + d] = acc;
}

// ---------------- packet-level GRU: grid (4,2); time=32, batch tile of 16 -------

__global__ __launch_bounds__(256) void pkt_gru_kernel(
    const float* __restrict__ pe,
    const bf16_t* __restrict__ wihf, const bf16_t* __restrict__ wihb,
    const bf16_t* __restrict__ whhf, const bf16_t* __restrict__ whhb,
    const float* __restrict__ bihf, const float* __restrict__ bihb,
    const float* __restrict__ bhhf, const float* __restrict__ bhhb,
    float* __restrict__ out2) {
  const int dir = blockIdx.y;
  const bf16_t* wih = dir ? wihb : wihf;
  const bf16_t* whh = dir ? whhb : whhf;
  const float* bih = dir ? bihb : bihf;
  const float* bhh = dir ? bhhb : bhhf;
  const int b0 = blockIdx.x * 16;

  __shared__ __align__(16) bf16_t x_bf[16 * 256];
  __shared__ __align__(16) bf16_t h_bf[16 * 128];
  __shared__ float xg[16 * 384];
  __shared__ float hg[16 * 384];

  const int tid = threadIdx.x;
  const int w = tid >> 5;
  const int m = tid >> 4;
  const int j0 = (tid & 15) * 8;
  const int d0 = (tid & 15) * 16;

  for (int i = tid; i < 1024; i += 256) ((unsigned*)h_bf)[i] = 0u;

  float bxr[8], bxz[8], bxn[8], bhr[8], bhz[8], bhn[8], hreg[8];
#pragma unroll
  for (int i = 0; i < 8; ++i) {
    bxr[i] = bih[j0 + i]; bxz[i] = bih[128 + j0 + i]; bxn[i] = bih[256 + j0 + i];
    bhr[i] = bhh[j0 + i]; bhz[i] = bhh[128 + j0 + i]; bhn[i] = bhh[256 + j0 + i];
    hreg[i] = 0.f;
  }

  for (int t = 0; t < 32; ++t) {
    const int tt = dir ? (31 - t) : t;
    {  // stage x (f32 -> bf16) for this step
      const float* src = pe + ((size_t)tt * 64 + b0 + m) * 256 + d0;
#pragma unroll
      for (int k = 0; k < 16; ++k) x_bf[m * 256 + d0 + k] = (bf16_t)src[k];
    }
    __syncthreads();

    v8f x0 = v8f_zero(), x1 = v8f_zero(), x2 = v8f_zero();
    v8f h0 = v8f_zero(), h1 = v8f_zero(), h2 = v8f_zero();
#pragma unroll
    for (int s = 0; s < 8; ++s) {
      v16bf a = load_fragA(x_bf, 256, s * 32);
      x0 = wmma_bf16(a, load_fragB(wih, 256, (w * 3 + 0) * 16, s * 32), x0);
      x1 = wmma_bf16(a, load_fragB(wih, 256, (w * 3 + 1) * 16, s * 32), x1);
      x2 = wmma_bf16(a, load_fragB(wih, 256, (w * 3 + 2) * 16, s * 32), x2);
    }
#pragma unroll
    for (int s = 0; s < 4; ++s) {
      v16bf a = load_fragA(h_bf, 128, s * 32);
      h0 = wmma_bf16(a, load_fragB(whh, 128, (w * 3 + 0) * 16, s * 32), h0);
      h1 = wmma_bf16(a, load_fragB(whh, 128, (w * 3 + 1) * 16, s * 32), h1);
      h2 = wmma_bf16(a, load_fragB(whh, 128, (w * 3 + 2) * 16, s * 32), h2);
    }
    store_fragD(xg, 384, (w * 3 + 0) * 16, x0);
    store_fragD(xg, 384, (w * 3 + 1) * 16, x1);
    store_fragD(xg, 384, (w * 3 + 2) * 16, x2);
    store_fragD(hg, 384, (w * 3 + 0) * 16, h0);
    store_fragD(hg, 384, (w * 3 + 1) * 16, h1);
    store_fragD(hg, 384, (w * 3 + 2) * 16, h2);
    __syncthreads();

    float* orow = out2 + ((size_t)tt * 64 + b0 + m) * 256 + dir * 128;
#pragma unroll
    for (int i = 0; i < 8; ++i) {
      const int j = j0 + i;
      float xr = xg[m * 384 + j] + bxr[i];
      float xz = xg[m * 384 + 128 + j] + bxz[i];
      float xn = xg[m * 384 + 256 + j] + bxn[i];
      float hr = hg[m * 384 + j] + bhr[i];
      float hz = hg[m * 384 + 128 + j] + bhz[i];
      float hn = hg[m * 384 + 256 + j] + bhn[i];
      float r = sigm(xr + hr), z = sigm(xz + hz);
      float nn = tanh_f(xn + r * hn);
      float h = (1.f - z) * nn + z * hreg[i];
      hreg[i] = h;
      h_bf[m * 128 + j] = (bf16_t)h;
      orow[j] = h;
    }
    __syncthreads();
  }
}

// ---------------- packet attention + softmax over b + classifier ----------------

__global__ __launch_bounds__(256) void final_kernel(
    const float* __restrict__ out2, const float* __restrict__ pa_w1,
    const float* __restrict__ pa_b1, const float* __restrict__ pa_w2,
    const float* __restrict__ pa_b2, const float* __restrict__ wf,
    const float* __restrict__ bfv, float* __restrict__ out) {
  const int p = blockIdx.x, tid = threadIdx.x;
  __shared__ float red[256];
  __shared__ float sc[64];
  __shared__ float fe[256];

  const int b = tid >> 2, h0 = (tid & 3) * 32;
  const float* row = out2 + (size_t)(p * 64 + b) * 256;
  float part = 0.f;
  for (int h = h0; h < h0 + 32; ++h) {
    const float* wrow = pa_w1 + h * 256;
    float dot = pa_b1[h];
#pragma unroll 8
    for (int d = 0; d < 256; ++d) dot += row[d] * wrow[d];
    part += tanh_f(dot) * pa_w2[h];
  }
  red[tid] = part;
  __syncthreads();
  if (tid < 64)
    sc[tid] = red[tid * 4] + red[tid * 4 + 1] + red[tid * 4 + 2] + red[tid * 4 + 3] +
              pa_b2[0];
  __syncthreads();

  if (tid < 64) red[tid] = sc[tid];
  __syncthreads();
  for (int o = 32; o >= 1; o >>= 1) {
    if (tid < o) red[tid] = fmaxf(red[tid], red[tid + o]);
    __syncthreads();
  }
  const float mx = red[0];
  __syncthreads();
  if (tid < 64) {
    const float e = __expf(sc[tid] - mx);
    red[tid] = e;
    sc[tid] = e;
  }
  __syncthreads();
  for (int o = 32; o >= 1; o >>= 1) {
    if (tid < o) red[tid] += red[tid + o];
    __syncthreads();
  }
  const float inv = 1.f / red[0];
  __syncthreads();

  {  // flow_emb[d] = sum_b out2[p][b][d] * attn2[b]
    const int d = tid;
    float acc = 0.f;
    for (int bb = 0; bb < 64; ++bb)
      acc += out2[(size_t)(p * 64 + bb) * 256 + d] * sc[bb];
    fe[d] = acc * inv;
  }
  __syncthreads();
  if (tid < 10) {
    float acc = bfv[tid];
#pragma unroll 8
    for (int d = 0; d < 256; ++d) acc += fe[d] * wf[tid * 256 + d];
    out[p * 10 + tid] = acc;
  }
}

// ---------------- launch ----------------

extern "C" void kernel_launch(void* const* d_in, const int* in_sizes, int n_in,
                              void* d_out, int out_size, void* d_ws, size_t ws_size,
                              hipStream_t stream) {
  (void)in_sizes; (void)n_in; (void)out_size; (void)ws_size;
  const long long* flow = (const long long*)d_in[0];
  const float* emb = (const float*)d_in[1];
  const float* b_wih_f = (const float*)d_in[2];
  const float* b_whh_f = (const float*)d_in[3];
  const float* b_bih_f = (const float*)d_in[4];
  const float* b_bhh_f = (const float*)d_in[5];
  const float* b_wih_b = (const float*)d_in[6];
  const float* b_whh_b = (const float*)d_in[7];
  const float* b_bih_b = (const float*)d_in[8];
  const float* b_bhh_b = (const float*)d_in[9];
  const float* p_wih_f = (const float*)d_in[10];
  const float* p_whh_f = (const float*)d_in[11];
  const float* p_bih_f = (const float*)d_in[12];
  const float* p_bhh_f = (const float*)d_in[13];
  const float* p_wih_b = (const float*)d_in[14];
  const float* p_whh_b = (const float*)d_in[15];
  const float* p_bih_b = (const float*)d_in[16];
  const float* p_bhh_b = (const float*)d_in[17];
  const float* ba_w1 = (const float*)d_in[18];
  const float* ba_b1 = (const float*)d_in[19];
  const float* ba_w2 = (const float*)d_in[20];
  const float* ba_b2 = (const float*)d_in[21];
  const float* pa_w1 = (const float*)d_in[22];
  const float* pa_b1 = (const float*)d_in[23];
  const float* pa_w2 = (const float*)d_in[24];
  const float* pa_b2 = (const float*)d_in[25];
  const float* wf = (const float*)d_in[26];
  const float* bfv = (const float*)d_in[27];

  char* ws = (char*)d_ws;
  bf16_t* table_f  = (bf16_t*)(ws + 0);        // 256*384 bf16
  bf16_t* table_b  = (bf16_t*)(ws + 196608);
  bf16_t* whhf_bf  = (bf16_t*)(ws + 393216);   // 384*128 bf16
  bf16_t* whhb_bf  = (bf16_t*)(ws + 491520);
  bf16_t* pwhhf_bf = (bf16_t*)(ws + 589824);
  bf16_t* pwhhb_bf = (bf16_t*)(ws + 688128);
  bf16_t* pwihf_bf = (bf16_t*)(ws + 786432);   // 384*256 bf16
  bf16_t* pwihb_bf = (bf16_t*)(ws + 983040);
  bf16_t* baw1_bf  = (bf16_t*)(ws + 1179648);  // 128*256 bf16
  float*  scores   = (float*)(ws + 1245184);   // 128*2048 f32
  float*  pemb     = (float*)(ws + 2293760);   // 2048*256 f32
  float*  out2     = (float*)(ws + 4390912);   // 2048*256 f32
  bf16_t* out_bytes = (bf16_t*)(ws + 6488064); // 128*2048*256 bf16 (134MB)

  prep_kernel<<<dim3(384, 9), 256, 0, stream>>>(
      emb, b_wih_f, b_bih_f, b_wih_b, b_bih_b, b_whh_f, b_whh_b, p_whh_f, p_whh_b,
      p_wih_f, p_wih_b, ba_w1, table_f, table_b, whhf_bf, whhb_bf, pwhhf_bf,
      pwhhb_bf, pwihf_bf, pwihb_bf, baw1_bf);

  byte_gru_kernel<<<dim3(128, 2), 256, 0, stream>>>(
      flow, table_f, table_b, whhf_bf, whhb_bf, b_bhh_f, b_bhh_b, out_bytes);

  byte_attn_kernel<<<dim3(16384), 256, 0, stream>>>(out_bytes, baw1_bf, ba_b1,
                                                    ba_w2, ba_b2, scores);

  byte_softmax_kernel<<<dim3(4096), 64, 0, stream>>>(scores);

  packet_emb_kernel<<<dim3(2048), 256, 0, stream>>>(out_bytes, scores, pemb);

  pkt_gru_kernel<<<dim3(4, 2), 256, 0, stream>>>(
      pemb, pwihf_bf, pwihb_bf, pwhhf_bf, pwhhb_bf, p_bih_f, p_bih_b, p_bhh_f,
      p_bhh_b, out2);

  final_kernel<<<dim3(32), 256, 0, stream>>>(out2, pa_w1, pa_b1, pa_w2, pa_b2, wf,
                                             bfv, (float*)d_out);
}
